// Model_39676907885250
// MI455X (gfx1250) — matrix operand order repaired and checked by reading, not compiled, not run on hardware
//
#include <hip/hip_runtime.h>
#include <hip/hip_bf16.h>

// ---------------------------------------------------------------------------
// MI455X (gfx1250) attention forward, algebraically folded:
//   S = sqrt(D) * ((Q@wk) @ x2^T + (Q@bk))      (K-projection folded into Q)
//   O = (softmax(S) @ x2) @ wv^T + bv           (V-projection folded after P)
// ~20.4 GFLOP total; HBM floor = 2 passes over x2 (536 MB ~= 23us @23.3TB/s).
// All matrix math via v_wmma_f32_16x16x32_f16 (wave32, f32 accumulate).
// k_scores uses a 1x8 wave layout so each x2 element is loaded exactly once
// per workgroup (B operand never duplicated across waves).
// ---------------------------------------------------------------------------

typedef __attribute__((ext_vector_type(16))) _Float16 v16h;
typedef __attribute__((ext_vector_type(8)))  _Float16 v8h;
typedef __attribute__((ext_vector_type(8)))  float    v8f;
typedef __attribute__((ext_vector_type(4)))  float    v4f;

#define DM    512
#define NB    32
#define LQ_   64
#define LKV_  4096
#define SQRTD 22.627416997969522f   // reference divides by D^-0.5 == multiply by sqrt(D)

static __device__ __forceinline__ v8f wmma16(v16h a, v16h b, v8f c) {
  return __builtin_amdgcn_wmma_f32_16x16x32_f16(false, a, false, b, (short)0, c,
                                                false, false);
}

// ---- fragment loaders (CDNA5 16x16x32 f16 VGPR striping) -------------------
// A (16x32, M x K), row-major f16 source. lane: m = m0+(lane&15);
// elems i -> k = k0 + 16*(lane>=16) + i (contiguous).
static __device__ __forceinline__ v16h frag_a_f16(const _Float16* __restrict__ src,
                                                  int ld, int m0, int k0) {
  const int lane = threadIdx.x & 31;
  const _Float16* p = src + (size_t)(m0 + (lane & 15)) * ld + k0 + ((lane >> 4) << 4);
  v8h lo = *(const v8h*)p;
  v8h hi = *(const v8h*)(p + 8);
  v16h r;
#pragma unroll
  for (int i = 0; i < 8; ++i) { r[i] = lo[i]; r[i + 8] = hi[i]; }
  return r;
}

static __device__ __forceinline__ v16h frag_a_f32(const float* __restrict__ src,
                                                  int ld, int m0, int k0) {
  const int lane = threadIdx.x & 31;
  const float* p = src + (size_t)(m0 + (lane & 15)) * ld + k0 + ((lane >> 4) << 4);
  v4f a0 = *(const v4f*)p;
  v4f a1 = *(const v4f*)(p + 4);
  v4f b0 = *(const v4f*)(p + 8);
  v4f b1 = *(const v4f*)(p + 12);
  v16h r;
#pragma unroll
  for (int i = 0; i < 4; ++i) {
    r[i]      = (_Float16)a0[i];
    r[i + 4]  = (_Float16)a1[i];
    r[i + 8]  = (_Float16)b0[i];
    r[i + 12] = (_Float16)b1[i];
  }
  return r;
}

// B (32x16, K x N) where memory is n-major: elem(k,n) = src[n*ld + k].
// lane: n = n0+(lane&15); elems i -> k = k0 + 16*(lane>=16) + i (contiguous).
static __device__ __forceinline__ v16h frag_b_nmajor_f32(const float* __restrict__ src,
                                                         int ld, int k0, int n0) {
  const int lane = threadIdx.x & 31;
  const float* p = src + (size_t)(n0 + (lane & 15)) * ld + k0 + ((lane >> 4) << 4);
  v16h r;
#pragma unroll
  for (int q = 0; q < 4; ++q) {
    v4f v = *(const v4f*)(p + 4 * q);
#pragma unroll
    for (int e = 0; e < 4; ++e) r[4 * q + e] = (_Float16)v[e];
  }
  return r;
}

// B fragment from an LDS tile stored n-major: ldsT[n][k], padded stride 40.
static __device__ __forceinline__ v16h frag_b_lds(const _Float16 (*ldsT)[40], int n0) {
  const int lane = threadIdx.x & 31;
  const _Float16* p = &ldsT[n0 + (lane & 15)][(lane >> 4) << 4];
  v8h lo = *(const v8h*)p;
  v8h hi = *(const v8h*)(p + 8);
  v16h r;
#pragma unroll
  for (int i = 0; i < 8; ++i) { r[i] = lo[i]; r[i + 8] = hi[i]; }
  return r;
}

// ---- 1) Q = x1 @ wq^T + bq  -> f16 [2048,512] ------------------------------
__global__ void __launch_bounds__(256) k_qproj(const float* __restrict__ x1,
                                               const float* __restrict__ wq,
                                               const float* __restrict__ bq,
                                               _Float16* __restrict__ Qh) {
  const int mblk = blockIdx.x * 64;    // rows of flattened [B*LQ]
  const int nblk = blockIdx.y * 128;   // output features
  const int wave = threadIdx.x >> 5, lane = threadIdx.x & 31;
  const int wm = (wave >> 2) * 32, wn = (wave & 3) * 32;
  v8f acc[2][2] = {};
  for (int k = 0; k < DM; k += 32) {
    v16h a0 = frag_a_f32(x1, DM, mblk + wm, k);
    v16h a1 = frag_a_f32(x1, DM, mblk + wm + 16, k);
    v16h b0 = frag_b_nmajor_f32(wq, DM, k, nblk + wn);        // wq[n*512+k]
    v16h b1 = frag_b_nmajor_f32(wq, DM, k, nblk + wn + 16);
    acc[0][0] = wmma16(a0, b0, acc[0][0]);
    acc[0][1] = wmma16(a0, b1, acc[0][1]);
    acc[1][0] = wmma16(a1, b0, acc[1][0]);
    acc[1][1] = wmma16(a1, b1, acc[1][1]);
  }
#pragma unroll
  for (int mi = 0; mi < 2; ++mi)
#pragma unroll
    for (int ni = 0; ni < 2; ++ni) {
      const int n = nblk + wn + ni * 16 + (lane & 15);
      const float bias = bq[n];
      const int m0 = mblk + wm + mi * 16 + ((lane >> 4) << 3);
#pragma unroll
      for (int r = 0; r < 8; ++r)
        Qh[(size_t)(m0 + r) * DM + n] = (_Float16)(acc[mi][ni][r] + bias);
    }
}

// ---- 2) c[row] = Q[row,:] . bk  -------------------------------------------
__global__ void __launch_bounds__(256) k_cvec(const _Float16* __restrict__ Qh,
                                              const float* __restrict__ bk,
                                              float* __restrict__ cbuf) {
  const int row  = blockIdx.x * 8 + (threadIdx.x >> 5);
  const int lane = threadIdx.x & 31;
  float s = 0.f;
  for (int k = lane; k < DM; k += 32)
    s += (float)Qh[(size_t)row * DM + k] * bk[k];
#pragma unroll
  for (int off = 16; off > 0; off >>= 1) s += __shfl_xor(s, off, 32);
  if (lane == 0) cbuf[row] = s;
}

// ---- 3) Q' = Q @ wk  (wk NOT transposed -> LDS transpose staging) ----------
__global__ void __launch_bounds__(256) k_qk_fold(const _Float16* __restrict__ Qh,
                                                 const float* __restrict__ wk,
                                                 _Float16* __restrict__ Qph) {
  __shared__ _Float16 ldsT[128][40];   // [n_local(d)][k_local(e)], padded
  const int mblk = blockIdx.x * 64;
  const int nblk = blockIdx.y * 128;
  const int wave = threadIdx.x >> 5, lane = threadIdx.x & 31;
  const int wm = (wave >> 2) * 32, wn = (wave & 3) * 32;
  v8f acc[2][2] = {};
  for (int kb = 0; kb < DM; kb += 32) {
    {  // stage wk[kb..kb+31][nblk..nblk+127] transposed into LDS as f16
      const int t = threadIdx.x;
      const int row = t >> 3;            // k local 0..31
      const int col = (t & 7) << 4;      // n local 0..112 step 16
      const float* src = wk + (size_t)(kb + row) * DM + nblk + col;
#pragma unroll
      for (int q = 0; q < 4; ++q) {
        v4f v = *(const v4f*)(src + 4 * q);
#pragma unroll
        for (int e = 0; e < 4; ++e) ldsT[col + 4 * q + e][row] = (_Float16)v[e];
      }
    }
    __syncthreads();
    v16h a0 = frag_a_f16(Qh, DM, mblk + wm, kb);
    v16h a1 = frag_a_f16(Qh, DM, mblk + wm + 16, kb);
    v16h b0 = frag_b_lds(ldsT, wn);
    v16h b1 = frag_b_lds(ldsT, wn + 16);
    acc[0][0] = wmma16(a0, b0, acc[0][0]);
    acc[0][1] = wmma16(a0, b1, acc[0][1]);
    acc[1][0] = wmma16(a1, b0, acc[1][0]);
    acc[1][1] = wmma16(a1, b1, acc[1][1]);
    __syncthreads();
  }
#pragma unroll
  for (int mi = 0; mi < 2; ++mi)
#pragma unroll
    for (int ni = 0; ni < 2; ++ni) {
      const int n = nblk + wn + ni * 16 + (lane & 15);
      const int m0 = mblk + wm + mi * 16 + ((lane >> 4) << 3);
#pragma unroll
      for (int r = 0; r < 8; ++r)
        Qph[(size_t)(m0 + r) * DM + n] = (_Float16)acc[mi][ni][r];
    }
}

// ---- 4) S = sqrt(D) * (Q' @ x2^T + c)  -> f32 [B,64,4096] ------------------
// 1x8 wave layout: each wave owns a distinct 16-wide kv strip -> the streamed
// x2 operand is loaded exactly once per workgroup. A (Q', 64KB/batch) is
// re-read by all 8 waves but is WGP$/L2 resident.
__global__ void __launch_bounds__(256) k_scores(const _Float16* __restrict__ Qph,
                                                const float* __restrict__ x2,
                                                const float* __restrict__ cbuf,
                                                float* __restrict__ S) {
  const int b = blockIdx.x;
  const int kvblk = blockIdx.y * 128;
  const int wave = threadIdx.x >> 5, lane = threadIdx.x & 31;
  const int wn = wave * 16;            // 8 waves x 16 kv columns
  const _Float16* A = Qph + (size_t)b * LQ_ * DM;
  const float*    X = x2  + (size_t)b * LKV_ * DM;   // x2^T: n=kv rows, k=d contiguous
  v8f acc[4] = {};
  for (int k = 0; k < DM; k += 32) {
    if (k + 32 < DM)  // prefetch next K-step of the streaming x2 line
      __builtin_prefetch(X + (size_t)(kvblk + wn + (lane & 15)) * DM + k + 32, 0, 1);
    v16h bf = frag_b_nmajor_f32(X, DM, k, kvblk + wn);
#pragma unroll
    for (int mi = 0; mi < 4; ++mi) {
      v16h af = frag_a_f16(A, DM, mi * 16, k);
      acc[mi] = wmma16(af, bf, acc[mi]);
    }
  }
  const int n = kvblk + wn + (lane & 15);
#pragma unroll
  for (int mi = 0; mi < 4; ++mi) {
    const int m0 = mi * 16 + ((lane >> 4) << 3);
#pragma unroll
    for (int r = 0; r < 8; ++r) {
      const int q = m0 + r;
      S[((size_t)b * LQ_ + q) * LKV_ + n] =
          SQRTD * (acc[mi][r] + cbuf[b * LQ_ + q]);
    }
  }
}

// ---- 5) row softmax over 4096, write P as f16 ------------------------------
__global__ void __launch_bounds__(256) k_softmax(const float* __restrict__ S,
                                                 _Float16* __restrict__ Ph) {
  const size_t row = blockIdx.x;
  const float* s = S + row * LKV_;
  _Float16* p = Ph + row * LKV_;
  __shared__ float red[256];
  const int t = threadIdx.x;
  float vals[16];
  float m = -3.4e38f;
#pragma unroll
  for (int i = 0; i < 16; ++i) { vals[i] = s[t + i * 256]; m = fmaxf(m, vals[i]); }
  red[t] = m; __syncthreads();
  for (int o = 128; o > 0; o >>= 1) {
    if (t < o) red[t] = fmaxf(red[t], red[t + o]);
    __syncthreads();
  }
  m = red[0]; __syncthreads();
  float sum = 0.f;
#pragma unroll
  for (int i = 0; i < 16; ++i) { vals[i] = __expf(vals[i] - m); sum += vals[i]; }
  red[t] = sum; __syncthreads();
  for (int o = 128; o > 0; o >>= 1) {
    if (t < o) red[t] += red[t + o];
    __syncthreads();
  }
  const float inv = 1.0f / red[0];
#pragma unroll
  for (int i = 0; i < 16; ++i) p[t + i * 256] = (_Float16)(vals[i] * inv);
}

// ---- 6) T_partial[chunk] = P[:, chunk] @ x2[chunk, :]  (split-K) -----------
__global__ void __launch_bounds__(256) k_pv(const _Float16* __restrict__ Ph,
                                            const float* __restrict__ x2,
                                            float* __restrict__ Tpart) {
  __shared__ _Float16 ldsT[128][40];   // x2 tile transposed: [d_local][kv_local]
  const int b = blockIdx.x;            // 32
  const int chunk = blockIdx.y;        // 8 KV chunks of 512
  const int nblk = blockIdx.z * 128;   // d
  const int wave = threadIdx.x >> 5, lane = threadIdx.x & 31;
  const int wm = (wave >> 2) * 32, wn = (wave & 3) * 32;
  const _Float16* A = Ph + (size_t)b * LQ_ * LKV_;
  const float*    X = x2 + (size_t)b * LKV_ * DM;
  v8f acc[2][2] = {};
  for (int kb = 0; kb < 512; kb += 32) {
    const int kv0 = chunk * 512 + kb;
    {  // stage x2[kv0..kv0+31][nblk..nblk+127] transposed into LDS as f16
      const int t = threadIdx.x;
      const int row = t >> 3;
      const int col = (t & 7) << 4;
      const float* src = X + (size_t)(kv0 + row) * DM + nblk + col;
#pragma unroll
      for (int q = 0; q < 4; ++q) {
        v4f v = *(const v4f*)(src + 4 * q);
#pragma unroll
        for (int e = 0; e < 4; ++e) ldsT[col + 4 * q + e][row] = (_Float16)v[e];
      }
    }
    __syncthreads();
    v16h a0 = frag_a_f16(A, LKV_, wm, kv0);
    v16h a1 = frag_a_f16(A, LKV_, wm + 16, kv0);
    v16h b0 = frag_b_lds(ldsT, wn);
    v16h b1 = frag_b_lds(ldsT, wn + 16);
    acc[0][0] = wmma16(a0, b0, acc[0][0]);
    acc[0][1] = wmma16(a0, b1, acc[0][1]);
    acc[1][0] = wmma16(a1, b0, acc[1][0]);
    acc[1][1] = wmma16(a1, b1, acc[1][1]);
    __syncthreads();
  }
#pragma unroll
  for (int mi = 0; mi < 2; ++mi)
#pragma unroll
    for (int ni = 0; ni < 2; ++ni) {
      const int n = nblk + wn + ni * 16 + (lane & 15);
      const int m0 = wm + mi * 16 + ((lane >> 4) << 3);
#pragma unroll
      for (int r = 0; r < 8; ++r)
        Tpart[((size_t)chunk * (NB * LQ_) + b * LQ_ + m0 + r) * DM + n] = acc[mi][ni][r];
    }
}

// ---- 7) reduce split-K partials, convert to f16 ----------------------------
__global__ void __launch_bounds__(256) k_reduceT(const float* __restrict__ Tpart,
                                                 _Float16* __restrict__ Tsumh) {
  const size_t i = (size_t)blockIdx.x * 256 + threadIdx.x;   // 1,048,576 elems
  float s = 0.f;
#pragma unroll
  for (int c = 0; c < 8; ++c) s += Tpart[(size_t)c * 1048576 + i];
  Tsumh[i] = (_Float16)s;
}

// ---- 8) O = T @ wv^T + bv  -> f32 output -----------------------------------
__global__ void __launch_bounds__(256) k_oproj(const _Float16* __restrict__ Tsumh,
                                               const float* __restrict__ wv,
                                               const float* __restrict__ bv,
                                               float* __restrict__ out) {
  const int mblk = blockIdx.x * 64;
  const int nblk = blockIdx.y * 128;
  const int wave = threadIdx.x >> 5, lane = threadIdx.x & 31;
  const int wm = (wave >> 2) * 32, wn = (wave & 3) * 32;
  v8f acc[2][2] = {};
  for (int k = 0; k < DM; k += 32) {
    v16h a0 = frag_a_f16(Tsumh, DM, mblk + wm, k);
    v16h a1 = frag_a_f16(Tsumh, DM, mblk + wm + 16, k);
    v16h b0 = frag_b_nmajor_f32(wv, DM, k, nblk + wn);
    v16h b1 = frag_b_nmajor_f32(wv, DM, k, nblk + wn + 16);
    acc[0][0] = wmma16(a0, b0, acc[0][0]);
    acc[0][1] = wmma16(a0, b1, acc[0][1]);
    acc[1][0] = wmma16(a1, b0, acc[1][0]);
    acc[1][1] = wmma16(a1, b1, acc[1][1]);
  }
#pragma unroll
  for (int mi = 0; mi < 2; ++mi)
#pragma unroll
    for (int ni = 0; ni < 2; ++ni) {
      const int n = nblk + wn + ni * 16 + (lane & 15);
      const float bias = bv[n];
      const int m0 = mblk + wm + mi * 16 + ((lane >> 4) << 3);
#pragma unroll
      for (int r = 0; r < 8; ++r)
        out[(size_t)(m0 + r) * DM + n] = acc[mi][ni][r] + bias;
    }
}

// ---------------------------------------------------------------------------
extern "C" void kernel_launch(void* const* d_in, const int* in_sizes, int n_in,
                              void* d_out, int out_size, void* d_ws, size_t ws_size,
                              hipStream_t stream) {
  const float* x1 = (const float*)d_in[0];   // [32,64,512]
  const float* x2 = (const float*)d_in[1];   // [32,4096,512]
  // d_in[2] = mask (unused by the reference forward)
  const float* wq = (const float*)d_in[3];
  const float* bq = (const float*)d_in[4];
  const float* wk = (const float*)d_in[5];
  const float* bk = (const float*)d_in[6];
  const float* wv = (const float*)d_in[7];
  const float* bv = (const float*)d_in[8];
  float* out = (float*)d_out;

  // workspace layout (~90 MiB total)
  char* ws = (char*)d_ws;
  _Float16* Qh    = (_Float16*)(ws + 0);                       //  2 MiB f16 [2048,512]
  _Float16* Qph   = (_Float16*)(ws + (size_t)2  * (1 << 20));  //  2 MiB f16 [2048,512]
  float*    cbuf  = (float*)   (ws + (size_t)4  * (1 << 20));  //  8 KiB f32 [2048]
  float*    S     = (float*)   (ws + (size_t)8  * (1 << 20));  // 32 MiB f32 [2048,4096]
  _Float16* Ph    = (_Float16*)(ws + (size_t)40 * (1 << 20));  // 16 MiB f16 [2048,4096]
  float*    Tpart = (float*)   (ws + (size_t)56 * (1 << 20));  // 32 MiB f32 [8,2048,512]
  _Float16* Tsumh = (_Float16*)(ws + (size_t)88 * (1 << 20));  //  2 MiB f16 [2048,512]

  k_qproj  <<<dim3(32, 4),     256, 0, stream>>>(x1, wq, bq, Qh);
  k_cvec   <<<256,             256, 0, stream>>>(Qh, bk, cbuf);
  k_qk_fold<<<dim3(32, 4),     256, 0, stream>>>(Qh, wk, Qph);
  k_scores <<<dim3(32, 32),    256, 0, stream>>>(Qph, x2, cbuf, S);
  k_softmax<<<2048,            256, 0, stream>>>(S, Ph);
  k_pv     <<<dim3(32, 8, 4),  256, 0, stream>>>(Ph, x2, Tpart);
  k_reduceT<<<4096,            256, 0, stream>>>(Tpart, Tsumh);
  k_oproj  <<<dim3(32, 4),     256, 0, stream>>>(Tsumh, wv, bv, out);
}
